// TorchMD_GN_50714973831855
// MI455X (gfx1250) — compile-verified, matrix operand order checked by reading, not run on hardware
//
#include <hip/hip_runtime.h>
#include <hip/hip_bf16.h>
#include <math.h>

typedef __attribute__((ext_vector_type(16))) _Float16 v16h;
typedef __attribute__((ext_vector_type(8)))  float    v8f;

#define HDIM 128
#define NT 8   // 128 / 16 output column tiles

// ---------------------------------------------------------------- WMMA core
__device__ __forceinline__ v8f wmma_f16(v16h a, v16h b, v8f c) {
  return __builtin_amdgcn_wmma_f32_16x16x32_f16(false, a, false, b, (short)0, c,
                                                false, false);
}

// A fragment (16x32 f16) from an f32 base + int32 offset; lane holds row
// (lane&15): elems 0..7 -> k = kstep*32 + (lane>>4)*8 + e, elems 8..15 -> +16.
__device__ __forceinline__ v16h load_afrag_f32(const float* __restrict__ base,
                                               int off, int kstep, int khalf,
                                               float scale) {
  int p = off + kstep * 32 + khalf * 8;
  v16h a;
#pragma unroll
  for (int e = 0; e < 8; ++e) a[e] = (_Float16)(base[p + e] * scale);
#pragma unroll
  for (int e = 0; e < 8; ++e) a[8 + e] = (_Float16)(base[p + 16 + e] * scale);
  return a;
}

__device__ __forceinline__ v16h load_afrag_h(const _Float16* __restrict__ base,
                                             int off, int kstep, int khalf) {
  int p = off + kstep * 32 + khalf * 8;
  v16h a;
#pragma unroll
  for (int e = 0; e < 8; ++e) a[e] = base[p + e];
#pragma unroll
  for (int e = 0; e < 8; ++e) a[8 + e] = base[p + 16 + e];
  return a;
}

// B fragments are pre-packed in LDS in lane-contiguous order: one 32B read.
__device__ __forceinline__ v16h load_bfrag(const _Float16* frag, int kstep,
                                           int ntile, int lane) {
  return *(const v16h*)(frag + ((((kstep << 3) + ntile) << 5) + lane) * 16);
}

// Convert a [K x 128] f32 row-major weight into f16 B fragments in LDS.
// Rows k >= Ksrc are zero-padded (used for R=50 -> K=64).
__device__ inline void fill_wfrags(_Float16* frag, const float* __restrict__ W,
                                   int Ksrc, int ksteps, int tid,
                                   int nthreads) {
  int total = ksteps * NT * 32 * 16;
  for (int idx = tid; idx < total; idx += nthreads) {
    int e = idx & 15;
    int lane = (idx >> 4) & 31;
    int ntile = (idx >> 9) & 7;
    int kstep = idx >> 12;
    int khalf = lane >> 4;
    int k = kstep * 32 + khalf * 8 + (e & 7) + ((e >> 3) << 4);
    int n = ntile * 16 + (lane & 15);
    float v = (k < Ksrc) ? W[k * HDIM + n] : 0.0f;
    frag[idx] = (_Float16)v;
  }
}

// --------------------------------------------------------- elementwise preps
__global__ void zero_kernel(float* p, int n) {
  int i = blockIdx.x * blockDim.x + threadIdx.x;
  if (i < n) p[i] = 0.0f;
}

__global__ void embed_kernel(const int* __restrict__ z,
                             const float* __restrict__ emb,
                             float* __restrict__ x, int N) {
  int idx = blockIdx.x * blockDim.x + threadIdx.x;
  if (idx >= N * HDIM) return;
  int n = idx >> 7, h = idx & 127;
  x[idx] = emb[z[n] * HDIM + h];
}

__global__ void edge_prep_kernel(const float* __restrict__ ew,
                                 const float* __restrict__ means,
                                 const float* __restrict__ betas,
                                 float* __restrict__ C,
                                 _Float16* __restrict__ attr, int E, int R) {
  int e = blockIdx.x * blockDim.x + threadIdx.x;
  if (e >= E) return;
  float d = ew[e];
  float cut = 0.5f * (cosf(d * 3.14159265358979323846f / 5.0f) + 1.0f) *
              (d < 5.0f ? 1.0f : 0.0f);
  C[e] = cut;
  float ed = expf(-d);
  int o = e * 64;
  for (int r = 0; r < R; ++r) {
    float t = ed - means[r];
    attr[o + r] = (_Float16)(cut * expf(-betas[r] * t * t));
  }
  for (int r = R; r < 64; ++r) attr[o + r] = (_Float16)0.0f;
}

__global__ void count_kernel(const int* __restrict__ ii, float* cnt, int E) {
  int e = blockIdx.x * blockDim.x + threadIdx.x;
  if (e < E) atomicAdd(&cnt[ii[e]], 1.0f);
}

__global__ void invcnt_kernel(const float* __restrict__ cnt, float* inv, int N) {
  int n = blockIdx.x * blockDim.x + threadIdx.x;
  if (n >= N) return;
  float c = cnt[n];
  inv[n] = (c > 0.0f) ? 1.0f / fmaxf(c, 1.0f) : 0.0f;
}

// ------------------------------------------------------------ node-side GEMM
// out[N x 128] = epilogue( [in0 | in1][N x 32*KSTEPS] @ W + bias )
template <int KSTEPS, int K0STEPS, bool HAS_BIAS, bool SILU, bool RESIDUAL,
          bool ROWSCALE>
__global__ __launch_bounds__(128) void gemm_node_kernel(
    const float* __restrict__ in0, const float* __restrict__ in1,
    const float* __restrict__ Wg, int Ksrc, const float* __restrict__ bias,
    const float* __restrict__ rowscale, const float* residual, float* out,
    int nrows) {
  __shared__ __align__(32) _Float16 wfrag[KSTEPS * NT * 32 * 16];
  fill_wfrags(wfrag, Wg, Ksrc, KSTEPS, threadIdx.x, blockDim.x);
  __syncthreads();
  const int wave = threadIdx.x >> 5, lane = threadIdx.x & 31;
  const int m = lane & 15, khalf = lane >> 4;
  const int rowblocks = nrows >> 4;
  const int stride0 = K0STEPS * 32;
  const int stride1 = (KSTEPS - K0STEPS) * 32;
  for (int rb = blockIdx.x * 4 + wave; rb < rowblocks; rb += gridDim.x * 4) {
    int arow = rb * 16 + m;
    float sc = ROWSCALE ? rowscale[arow] : 1.0f;
    int o0 = arow * stride0;
    int o1 = arow * stride1;
    v16h a[KSTEPS];
#pragma unroll
    for (int ks = 0; ks < KSTEPS; ++ks)
      a[ks] = (ks < K0STEPS)
                  ? load_afrag_f32(in0, o0, ks, khalf, sc)
                  : load_afrag_f32(in1, o1, ks - K0STEPS, khalf, sc);
    // D-layout epilogue: int32 base offset; r/nt become instruction imms.
    int obase = (rb * 16 + 8 * khalf) * HDIM + m;
#pragma unroll
    for (int nt = 0; nt < NT; ++nt) {
      v8f acc = {};
#pragma unroll
      for (int ks = 0; ks < KSTEPS; ++ks)
        acc = wmma_f16(a[ks], load_bfrag(wfrag, ks, nt, lane), acc);
      float bv = HAS_BIAS ? bias[nt * 16 + m] : 0.0f;
#pragma unroll
      for (int r = 0; r < 8; ++r) {
        float v = acc[r] + bv;
        if (SILU) v = v / (1.0f + expf(-v));
        if (RESIDUAL) v += residual[obase + r * HDIM + nt * 16];
        out[obase + r * HDIM + nt * 16] = v;
      }
    }
  }
}

// ------------------------------------------------------------ edge-side GEMM
// Per 16-edge tile: W = [silu(attr@w1+b1)@w2+b2] (TWO) or [attr@w1+b1] (dp),
// then scatter: outacc[i] += (W * C) * src[j], all fused in registers/LDS.
template <bool TWO>
__global__ __launch_bounds__(128) void edge_kernel(
    const _Float16* __restrict__ attr_h, const float* __restrict__ Cc,
    const int* __restrict__ jidx, const int* __restrict__ iidx,
    const float* __restrict__ w1, int k1src, const float* __restrict__ b1,
    const float* __restrict__ w2, const float* __restrict__ b2,
    const float* __restrict__ src, float* __restrict__ outacc, int nedges) {
  __shared__ __align__(32) _Float16 w1frag[2 * NT * 32 * 16];
  __shared__ __align__(32) _Float16 w2frag[TWO ? 4 * NT * 32 * 16 : 16];
  __shared__ __align__(32) _Float16 tbuf[TWO ? 4 * 16 * HDIM : 16];
  fill_wfrags(w1frag, w1, k1src, 2, threadIdx.x, blockDim.x);
  if (TWO) fill_wfrags(w2frag, w2, 128, 4, threadIdx.x, blockDim.x);
  __syncthreads();
  const int wave = threadIdx.x >> 5, lane = threadIdx.x & 31;
  const int m = lane & 15, khalf = lane >> 4;
  const int eblocks = nedges >> 4;
  _Float16* tw = tbuf + wave * 16 * HDIM;
  for (int eb = blockIdx.x * 4 + wave; eb < eblocks; eb += gridDim.x * 4) {
    int arow = eb * 16 + m;
    v16h a0 = load_afrag_h(attr_h, arow * 64, 0, khalf);
    v16h a1 = load_afrag_h(attr_h, arow * 64, 1, khalf);
    // Per-lane row metadata as int32 offsets: loads/atomics then use the
    // uniform SGPR base + 32-bit VGPR offset + immediate addressing mode.
    float cr[8];
    int so[8], oo[8];
#pragma unroll
    for (int r = 0; r < 8; ++r) {
      int gm = eb * 16 + r + 8 * khalf;
      cr[r] = Cc[gm];
      so[r] = jidx[gm] * HDIM + m;
      oo[r] = iidx[gm] * HDIM + m;
    }
#pragma unroll
    for (int nt = 0; nt < NT; ++nt) {
      v8f acc = {};
      acc = wmma_f16(a0, load_bfrag(w1frag, 0, nt, lane), acc);
      acc = wmma_f16(a1, load_bfrag(w1frag, 1, nt, lane), acc);
      float bv = b1[nt * 16 + m];
      if (TWO) {
#pragma unroll
        for (int r = 0; r < 8; ++r) {
          float v = acc[r] + bv;
          v = v / (1.0f + expf(-v));  // silu
          tw[(r + 8 * khalf) * HDIM + nt * 16 + m] = (_Float16)v;
        }
      } else {
#pragma unroll
        for (int r = 0; r < 8; ++r) {
          float w = (acc[r] + bv) * cr[r];
          float g = src[so[r] + nt * 16];
          atomicAdd(&outacc[oo[r] + nt * 16], w * g);
        }
      }
    }
    if (TWO) {
      v16h ta[4];
#pragma unroll
      for (int ks = 0; ks < 4; ++ks)
        ta[ks] = load_afrag_h(tw, m * HDIM, ks, khalf);
#pragma unroll
      for (int nt = 0; nt < NT; ++nt) {
        v8f acc = {};
#pragma unroll
        for (int ks = 0; ks < 4; ++ks)
          acc = wmma_f16(ta[ks], load_bfrag(w2frag, ks, nt, lane), acc);
        float bv = b2[nt * 16 + m];
#pragma unroll
        for (int r = 0; r < 8; ++r) {
          float w = (acc[r] + bv) * cr[r];
          float g = src[so[r] + nt * 16];
          atomicAdd(&outacc[oo[r] + nt * 16], w * g);
        }
      }
    }
  }
}

// ------------------------------------------------------------------- driver
extern "C" void kernel_launch(void* const* d_in, const int* in_sizes, int n_in,
                              void* d_out, int out_size, void* d_ws,
                              size_t ws_size, hipStream_t stream) {
  const int* z       = (const int*)d_in[0];
  const int* eidx    = (const int*)d_in[1];
  const float* ew    = (const float*)d_in[2];
  const float* emb   = (const float*)d_in[3];
  const float* means = (const float*)d_in[4];
  const float* betas = (const float*)d_in[5];
  const float* dp_w  = (const float*)d_in[6];
  const float* dp_b  = (const float*)d_in[7];
  const float* comb_w = (const float*)d_in[8];
  const float* comb_b = (const float*)d_in[9];
  const float* mlp_w1 = (const float*)d_in[10];
  const float* mlp_b1 = (const float*)d_in[11];
  const float* mlp_w2 = (const float*)d_in[12];
  const float* mlp_b2 = (const float*)d_in[13];
  const float* lin1_w = (const float*)d_in[14];
  const float* lin2_w = (const float*)d_in[15];
  const float* lin2_b = (const float*)d_in[16];
  const float* lin_w  = (const float*)d_in[17];
  const float* lin_b  = (const float*)d_in[18];

  const int N = in_sizes[0];
  const int E = in_sizes[1] / 2;
  const int R = in_sizes[4];
  const int L = in_sizes[11] / HDIM;
  const int* jj = eidx;
  const int* ii = eidx + E;

  char* ws = (char*)d_ws;
  size_t off = 0;
  auto alloc = [&](size_t bytes) {
    char* p = ws + off;
    off += (bytes + 255) & ~(size_t)255;
    return p;
  };
  float* x0   = (float*)alloc((size_t)N * HDIM * 4);  // emb x / per-layer hpre
  float* xcur = (float*)alloc((size_t)N * HDIM * 4);
  float* aggb = (float*)alloc((size_t)N * HDIM * 4);  // agg / per-layer hagg
  float* hmid = (float*)alloc((size_t)N * HDIM * 4);
  float* cnt  = (float*)alloc((size_t)N * 4);
  float* inv  = (float*)alloc((size_t)N * 4);
  float* Cb   = (float*)alloc((size_t)E * 4);
  _Float16* attr = (_Float16*)alloc((size_t)E * 64 * 2);
  (void)ws_size; (void)n_in;

  const int nh = N * HDIM;
  zero_kernel<<<(nh + 255) / 256, 256, 0, stream>>>(aggb, nh);
  zero_kernel<<<(N + 255) / 256, 256, 0, stream>>>(cnt, N);
  embed_kernel<<<(nh + 255) / 256, 256, 0, stream>>>(z, emb, x0, N);
  edge_prep_kernel<<<(E + 255) / 256, 256, 0, stream>>>(ew, means, betas, Cb,
                                                        attr, E, R);
  count_kernel<<<(E + 255) / 256, 256, 0, stream>>>(ii, cnt, E);
  invcnt_kernel<<<(N + 255) / 256, 256, 0, stream>>>(cnt, inv, N);

  // Grid-stride with modest grids so the per-WG LDS weight-fragment build is
  // amortized over several 16-row tiles per wave.
  const int eblk = E / 16, nblk = N / 16;
  int egrid = (eblk + 3) / 4;  if (egrid > 1024) egrid = 1024;
  int ngrid = (nblk + 3) / 4;  if (ngrid > 160)  ngrid = 160;

  // agg = segment_sum(x[j] * (attr@dp_w + dp_b)*C, i)
  edge_kernel<false><<<egrid, 128, 0, stream>>>(attr, Cb, jj, ii, dp_w, R, dp_b,
                                                (const float*)0,
                                                (const float*)0, x0, aggb, E);
  // x = concat(x, agg) @ comb_w + comb_b
  gemm_node_kernel<8, 4, true, false, false, false><<<ngrid, 128, 0, stream>>>(
      x0, aggb, comb_w, 256, comb_b, nullptr, nullptr, xcur, N);

  for (int l = 0; l < L; ++l) {
    // hpre = x @ lin1_w[l]
    gemm_node_kernel<4, 4, false, false, false, false>
        <<<ngrid, 128, 0, stream>>>(xcur, nullptr,
                                    lin1_w + (size_t)l * HDIM * HDIM, 128,
                                    nullptr, nullptr, nullptr, x0, N);
    zero_kernel<<<(nh + 255) / 256, 256, 0, stream>>>(aggb, nh);
    // hagg = segment_sum(hpre[j] * W, i)
    edge_kernel<true><<<egrid, 128, 0, stream>>>(
        attr, Cb, jj, ii, mlp_w1 + (size_t)l * R * HDIM, R,
        mlp_b1 + (size_t)l * HDIM, mlp_w2 + (size_t)l * HDIM * HDIM,
        mlp_b2 + (size_t)l * HDIM, x0, aggb, E);
    // hmid = silu((hagg * inv_cnt) @ lin2_w[l] + lin2_b[l])
    gemm_node_kernel<4, 4, true, true, false, true><<<ngrid, 128, 0, stream>>>(
        aggb, nullptr, lin2_w + (size_t)l * HDIM * HDIM, 128,
        lin2_b + (size_t)l * HDIM, inv, nullptr, hmid, N);
    // x = x + (hmid @ lin_w[l] + lin_b[l])
    float* outp = (l == L - 1) ? (float*)d_out : xcur;
    gemm_node_kernel<4, 4, true, false, true, false><<<ngrid, 128, 0, stream>>>(
        hmid, nullptr, lin_w + (size_t)l * HDIM * HDIM, 128,
        lin_b + (size_t)l * HDIM, nullptr, xcur, outp, N);
  }
}
